// MaskedMultiHeadedSelfAttention_7069516169387
// MI455X (gfx1250) — matrix operand order repaired and checked by reading, not compiled, not run on hardware
//
#include <hip/hip_runtime.h>
#include <hip/hip_bf16.h>

namespace {
constexpr int kB  = 2;
constexpr int kS  = 2048;
constexpr int kE  = 1024;
constexpr int kH  = 16;
constexpr int kDH = 64;
constexpr int kBS = kB * kS;
} // namespace

typedef __attribute__((ext_vector_type(16))) __bf16 v16bf;
typedef __attribute__((ext_vector_type(8)))  float  v8f;

union Frag16 {
  v16bf v;
  uint4 q[2];
};
static_assert(sizeof(Frag16) == 32, "Frag16 must be 32 bytes");

__device__ __forceinline__ v8f wmma_bf16(v16bf a, v16bf b, v8f c) {
  // 8 args: (neg_a, A, neg_b, B, c_mod, C, reuse_a, reuse_b)
  return __builtin_amdgcn_wmma_f32_16x16x32_bf16(false, a, false, b, (short)0, c,
                                                 false, false);
}

__device__ __forceinline__ v8f vzero8() {
  v8f c;
#pragma unroll
  for (int i = 0; i < 8; ++i) c[i] = 0.0f;
  return c;
}

// Async global->LDS copy, 16B per lane (CDNA5, tracked by ASYNCcnt).
__device__ __forceinline__ void async_b128(unsigned lds_byte_off,
                                           const void* gptr) {
  asm volatile("global_load_async_to_lds_b128 %0, %1, off"
               :: "v"(lds_byte_off), "v"(gptr)
               : "memory");
}

// ---------------- conversion / layout kernels ----------------

__global__ void cvt_f32_bf16(const float* __restrict__ in,
                             __bf16* __restrict__ out, int n) {
  int i = blockIdx.x * blockDim.x + threadIdx.x;
  if (i < n) out[i] = (__bf16)in[i];
}

// w: [H, E, DH] (f32) -> wT: [H, DH, E] (bf16)  (N-major for B-fragment loads)
__global__ void cvt_transpose_head_w(const float* __restrict__ w,
                                     __bf16* __restrict__ wT) {
  int i = blockIdx.x * blockDim.x + threadIdx.x;
  if (i >= kH * kE * kDH) return;
  int d = i % kDH;
  int e = (i / kDH) % kE;
  int h = i / (kDH * kE);
  wT[((size_t)h * kDH + d) * kE + e] = (__bf16)w[i];
}

// wo: [E, E] (f32) -> woT[n*E + k] = wo[k*E + n] (bf16)
__global__ void cvt_transpose_wo(const float* __restrict__ wo,
                                 __bf16* __restrict__ woT) {
  int i = blockIdx.x * blockDim.x + threadIdx.x;
  if (i >= kE * kE) return;
  int n = i % kE;
  int k = i / kE;
  woT[(size_t)n * kE + k] = (__bf16)wo[i];
}

// ---------------- fragment loaders (global, register double-buffer) --------

__device__ __forceinline__ void load_ab(Frag16& a0, Frag16& a1, Frag16 (&b)[4],
                                        const __bf16* arow0, const __bf16* arow1,
                                        const __bf16* wb, int k0, int half,
                                        int l16) {
  a0.q[0] = *(const uint4*)(arow0 + k0 + half * 8);
  a0.q[1] = *(const uint4*)(arow0 + k0 + half * 8 + 16);
  a1.q[0] = *(const uint4*)(arow1 + k0 + half * 8);
  a1.q[1] = *(const uint4*)(arow1 + k0 + half * 8 + 16);
#pragma unroll
  for (int j = 0; j < 4; ++j) {
    const __bf16* bp = wb + (size_t)(j * 16 + l16) * kE + k0 + half * 8;
    b[j].q[0] = *(const uint4*)(bp);
    b[j].q[1] = *(const uint4*)(bp + 16);
  }
}

// ---------------- QKV projection: one wave = 32 rows x 64 dh ----------------
__global__ __launch_bounds__(32) void qkv_proj(
    const __bf16* __restrict__ xb,   // [BS, E]
    const __bf16* __restrict__ wT,   // [H, DH, E]
    const float*  __restrict__ bias, // [H, DH]
    __bf16* __restrict__ outp, int transposed, float scale) {
  const int lane = threadIdx.x & 31;
  const int half = lane >> 4;
  const int l16  = lane & 15;
  const int rowbase = blockIdx.x * 32;
  const int h = blockIdx.y;

  const __bf16* arow0 = xb + (size_t)(rowbase + l16) * kE;
  const __bf16* arow1 = xb + (size_t)(rowbase + 16 + l16) * kE;
  const __bf16* wb    = wT + (size_t)h * kDH * kE;

  v8f acc0[4], acc1[4];
#pragma unroll
  for (int j = 0; j < 4; ++j) { acc0[j] = vzero8(); acc1[j] = vzero8(); }

  Frag16 aA0, aA1, bA[4], aB0, aB1, bB[4];
  load_ab(aA0, aA1, bA, arow0, arow1, wb, 0, half, l16);
  for (int k0 = 0; k0 < kE; k0 += 64) {
    load_ab(aB0, aB1, bB, arow0, arow1, wb, k0 + 32, half, l16);
#pragma unroll
    for (int j = 0; j < 4; ++j) {
      acc0[j] = wmma_bf16(aA0.v, bA[j].v, acc0[j]);
      acc1[j] = wmma_bf16(aA1.v, bA[j].v, acc1[j]);
    }
    if (k0 + 64 < kE)
      load_ab(aA0, aA1, bA, arow0, arow1, wb, k0 + 64, half, l16);
#pragma unroll
    for (int j = 0; j < 4; ++j) {
      acc0[j] = wmma_bf16(aB0.v, bB[j].v, acc0[j]);
      acc1[j] = wmma_bf16(aB1.v, bB[j].v, acc1[j]);
    }
  }

  const int b = rowbase >> 11;           // S == 2048
  const int sbase = rowbase & (kS - 1);
#pragma unroll
  for (int j = 0; j < 4; ++j) {
    const int d = j * 16 + l16;
    const float bv = bias[h * kDH + d];
#pragma unroll
    for (int g = 0; g < 2; ++g) {
#pragma unroll
      for (int r = 0; r < 8; ++r) {
        const int seq = sbase + g * 16 + r + half * 8;
        const float val = ((g ? acc1[j][r] : acc0[j][r]) + bv) * scale;
        if (transposed)
          outp[((size_t)(b * kH + h) * kDH + d) * kS + seq] = (__bf16)val;
        else
          outp[((size_t)(b * kH + h) * kS + seq) * kDH + d] = (__bf16)val;
      }
    }
  }
}

// ---------------- flash attention (4 waves/block, async LDS staging) -------

// Stage one 32-key K tile (32x64 bf16) + V tile (64x32 bf16) into LDS.
// 128 threads x 4 async B128 ops = 8 KB. ASYNCcnt += 4 per wave.
__device__ __forceinline__ void stage_kv(unsigned koff, unsigned voff,
                                         const __bf16* kg_bh,
                                         const __bf16* vtg_bh, int kt,
                                         int tid) {
  {
    const int c0 = tid;            // K chunks 0..127
    const int c1 = tid + 128;      // K chunks 128..255
    async_b128(koff + c0 * 16,
               kg_bh + (size_t)(kt + (c0 >> 3)) * kDH + (c0 & 7) * 8);
    async_b128(koff + c1 * 16,
               kg_bh + (size_t)(kt + (c1 >> 3)) * kDH + (c1 & 7) * 8);
  }
  {
    const int c0 = tid;            // V chunks (4 per d-row)
    const int c1 = tid + 128;
    async_b128(voff + c0 * 16,
               vtg_bh + (size_t)(c0 >> 2) * kS + kt + (c0 & 3) * 8);
    async_b128(voff + c1 * 16,
               vtg_bh + (size_t)(c1 >> 2) * kS + kt + (c1 & 3) * 8);
  }
}

__device__ __forceinline__ void attn_step(
    const __bf16* __restrict__ kst,  // LDS [32 keys][64 d]
    const __bf16* __restrict__ vst,  // LDS [64 d][32 keys]
    const Frag16& qa0, const Frag16& qa1,
    const int* __restrict__ mrowp, int kt, int half, int l16,
    float (&mrow)[8], float (&lrow)[8], v8f (&oacc)[4],
    __bf16* __restrict__ plds) {
  // ---- K fragments from LDS ----
  Frag16 kb00, kb01, kb10, kb11;
  {
    const __bf16* kr = kst + l16 * 64 + half * 8;
    kb00.q[0] = *(const uint4*)(kr);
    kb00.q[1] = *(const uint4*)(kr + 16);
    kb01.q[0] = *(const uint4*)(kr + 32);
    kb01.q[1] = *(const uint4*)(kr + 48);
  }
  {
    const __bf16* kr = kst + (16 + l16) * 64 + half * 8;
    kb10.q[0] = *(const uint4*)(kr);
    kb10.q[1] = *(const uint4*)(kr + 16);
    kb11.q[0] = *(const uint4*)(kr + 32);
    kb11.q[1] = *(const uint4*)(kr + 48);
  }

  // ---- scores: 16 queries x 32 keys (two 16x16 C tiles) ----
  v8f sc[2];
  {
    v8f c = vzero8();
    c = wmma_bf16(qa0.v, kb00.v, c);
    c = wmma_bf16(qa1.v, kb01.v, c);
    sc[0] = c;
  }
  {
    v8f c = vzero8();
    c = wmma_bf16(qa0.v, kb10.v, c);
    c = wmma_bf16(qa1.v, kb11.v, c);
    sc[1] = c;
  }

  // ---- key-padding mask ----
  const int mv0 = mrowp[kt + l16];
  const int mv1 = mrowp[kt + 16 + l16];
#pragma unroll
  for (int r = 0; r < 8; ++r) {
    if (mv0 == 0) sc[0][r] = -1.0e9f;
    if (mv1 == 0) sc[1][r] = -1.0e9f;
  }

  // ---- online softmax (a row lives on one 16-lane half) ----
  float tmax[8];
#pragma unroll
  for (int r = 0; r < 8; ++r) tmax[r] = fmaxf(sc[0][r], sc[1][r]);
#pragma unroll
  for (int off = 8; off >= 1; off >>= 1) {
#pragma unroll
    for (int r = 0; r < 8; ++r)
      tmax[r] = fmaxf(tmax[r], __shfl_xor(tmax[r], off, 32));
  }

  float alpha[8], psum[8];
#pragma unroll
  for (int r = 0; r < 8; ++r) {
    const float mnew = fmaxf(mrow[r], tmax[r]);
    alpha[r] = __expf(mrow[r] - mnew);
    mrow[r] = mnew;
    const float p0 = __expf(sc[0][r] - mnew);
    const float p1 = __expf(sc[1][r] - mnew);
    sc[0][r] = p0;
    sc[1][r] = p1;
    psum[r] = p0 + p1;
  }
#pragma unroll
  for (int off = 8; off >= 1; off >>= 1) {
#pragma unroll
    for (int r = 0; r < 8; ++r) psum[r] += __shfl_xor(psum[r], off, 32);
  }
#pragma unroll
  for (int r = 0; r < 8; ++r) lrow[r] = lrow[r] * alpha[r] + psum[r];
#pragma unroll
  for (int j = 0; j < 4; ++j) {
#pragma unroll
    for (int r = 0; r < 8; ++r) oacc[j][r] *= alpha[r];
  }

  // ---- P (C-layout) -> LDS -> A-fragment layout ----
#pragma unroll
  for (int nt = 0; nt < 2; ++nt) {
#pragma unroll
    for (int r = 0; r < 8; ++r)
      plds[(r + half * 8) * 32 + nt * 16 + l16] = (__bf16)sc[nt][r];
  }
  asm volatile("s_wait_dscnt 0" ::: "memory");  // CDNA5 split DS counter

  Frag16 pa;
  pa.q[0] = *(const uint4*)(&plds[l16 * 32 + half * 8]);
  pa.q[1] = *(const uint4*)(&plds[l16 * 32 + half * 8 + 16]);

  // ---- O += P @ V (V fragments from LDS) ----
#pragma unroll
  for (int j = 0; j < 4; ++j) {
    const __bf16* vr = vst + (j * 16 + l16) * 32 + half * 8;
    Frag16 vb;
    vb.q[0] = *(const uint4*)(vr);
    vb.q[1] = *(const uint4*)(vr + 16);
    oacc[j] = wmma_bf16(pa.v, vb.v, oacc[j]);
  }
}

__global__ __launch_bounds__(128) void attn_flash(
    const __bf16* __restrict__ qg,   // [B,H,S,DH], pre-scaled by 1/sqrt(DH)
    const __bf16* __restrict__ kg,   // [B,H,S,DH]
    const __bf16* __restrict__ vtg,  // [B,H,DH,S]
    const int*    __restrict__ mask, // [B,S]
    __bf16* __restrict__ ob) {       // [B,S,E] == [B,S,H,DH]
  // LDS: double-buffered K (2x4KB) + V (2x4KB) + per-wave P (4x1KB) = 20KB
  __shared__ __align__(16) __bf16 smem[2 * 2048 + 2 * 2048 + 4 * 512];

  const int tid  = threadIdx.x;
  const int wid  = tid >> 5;
  const int lane = tid & 31;
  const int half = lane >> 4;
  const int l16  = lane & 15;
  const int h = blockIdx.y;
  const int b = blockIdx.z;

  __bf16* plds = smem + 8192 + wid * 512;
  // LDS aperture flat addresses carry the LDS offset in addr[31:0].
  const unsigned smem0 = (unsigned)(uintptr_t)(void*)smem;

  const size_t bh = (size_t)(b * kH + h) * kS * kDH;
  const __bf16* kg_bh  = kg + bh;
  const __bf16* vtg_bh = vtg + bh;
  const int mbase = blockIdx.x * 64 + wid * 16;  // 16-query tile per wave
  const __bf16* qrow = qg + bh + (size_t)(mbase + l16) * kDH;

  Frag16 qa0, qa1;  // A-fragments for K-dim 0..31 and 32..63
  qa0.q[0] = *(const uint4*)(qrow + half * 8);
  qa0.q[1] = *(const uint4*)(qrow + half * 8 + 16);
  qa1.q[0] = *(const uint4*)(qrow + 32 + half * 8);
  qa1.q[1] = *(const uint4*)(qrow + 32 + half * 8 + 16);

  float mrow[8], lrow[8];
  v8f oacc[4];
#pragma unroll
  for (int r = 0; r < 8; ++r) { mrow[r] = -3.0e38f; lrow[r] = 0.0f; }
#pragma unroll
  for (int j = 0; j < 4; ++j) oacc[j] = vzero8();

  const int* mrowp = mask + b * kS;

  // async double-buffered key loop (32 keys per step)
  // byte layout: K buffers at 0 / 4096; V buffers at 8192 / 12288
  stage_kv(smem0, smem0 + 8192u, kg_bh, vtg_bh, 0, tid);
  for (int kt = 0; kt < kS; kt += 32) {
    const int cur = (kt >> 5) & 1;
    const unsigned koff_nxt = smem0 + (cur ? 0u : 4096u);
    const unsigned voff_nxt = smem0 + 8192u + (cur ? 0u : 4096u);
    const __bf16* kst = smem + (cur ? 2048 : 0);
    const __bf16* vst = smem + 4096 + (cur ? 2048 : 0);
    if (kt + 32 < kS) {
      stage_kv(koff_nxt, voff_nxt, kg_bh, vtg_bh, kt + 32, tid);
      asm volatile("s_wait_asynccnt 0x4" ::: "memory");  // drain current tile
    } else {
      asm volatile("s_wait_asynccnt 0x0" ::: "memory");
    }
    __syncthreads();  // current tile visible to all waves
    attn_step(kst, vst, qa0, qa1, mrowp, kt, half, l16, mrow, lrow, oacc,
              plds);
    __syncthreads();  // all waves done reading before next overwrite
  }

  // ---- epilogue: normalize and write concat-head output ----
#pragma unroll
  for (int r = 0; r < 8; ++r) {
    const float inv = 1.0f / lrow[r];
    const int seq = mbase + r + half * 8;
    __bf16* orow = ob + ((size_t)b * kS + seq) * kE + h * kDH;
#pragma unroll
    for (int j = 0; j < 4; ++j)
      orow[j * 16 + l16] = (__bf16)(oacc[j][r] * inv);
  }
}

// ---------------- output projection: one wave = 32 rows x 64 cols ----------
__global__ __launch_bounds__(32) void out_proj(
    const __bf16* __restrict__ ob,   // [BS, E]
    const __bf16* __restrict__ woT,  // [E, E]: woT[n*E+k] = wo[k*E+n]
    const float*  __restrict__ bo,   // [E]
    float* __restrict__ out) {       // [BS, E] f32
  const int lane = threadIdx.x & 31;
  const int half = lane >> 4;
  const int l16  = lane & 15;
  const int rowbase = blockIdx.x * 32;
  const int nbase = blockIdx.y * 64;

  const __bf16* arow0 = ob + (size_t)(rowbase + l16) * kE;
  const __bf16* arow1 = ob + (size_t)(rowbase + 16 + l16) * kE;
  const __bf16* wb    = woT + (size_t)nbase * kE;

  v8f acc0[4], acc1[4];
#pragma unroll
  for (int j = 0; j < 4; ++j) { acc0[j] = vzero8(); acc1[j] = vzero8(); }

  Frag16 aA0, aA1, bA[4], aB0, aB1, bB[4];
  load_ab(aA0, aA1, bA, arow0, arow1, wb, 0, half, l16);
  for (int k0 = 0; k0 < kE; k0 += 64) {
    load_ab(aB0, aB1, bB, arow0, arow1, wb, k0 + 32, half, l16);
#pragma unroll
    for (int j = 0; j < 4; ++j) {
      acc0[j] = wmma_bf16(aA0.v, bA[j].v, acc0[j]);
      acc1[j] = wmma_bf16(aA1.v, bA[j].v, acc1[j]);
    }
    if (k0 + 64 < kE)
      load_ab(aA0, aA1, bA, arow0, arow1, wb, k0 + 64, half, l16);
#pragma unroll
    for (int j = 0; j < 4; ++j) {
      acc0[j] = wmma_bf16(aB0.v, bB[j].v, acc0[j]);
      acc1[j] = wmma_bf16(aB1.v, bB[j].v, acc1[j]);
    }
  }

#pragma unroll
  for (int j = 0; j < 4; ++j) {
    const int n = nbase + j * 16 + l16;
    const float bias = bo[n];
#pragma unroll
    for (int g = 0; g < 2; ++g) {
#pragma unroll
      for (int r = 0; r < 8; ++r) {
        const int row = rowbase + g * 16 + r + half * 8;
        out[(size_t)row * kE + n] = (g ? acc1[j][r] : acc0[j][r]) + bias;
      }
    }
  }
}

// ---------------- driver ----------------

extern "C" void kernel_launch(void* const* d_in, const int* in_sizes, int n_in,
                              void* d_out, int out_size, void* d_ws, size_t ws_size,
                              hipStream_t stream) {
  (void)in_sizes; (void)n_in; (void)out_size; (void)ws_size;
  const float* x    = (const float*)d_in[0];
  const int*   mask = (const int*)d_in[1];
  const float* wq   = (const float*)d_in[2];
  const float* bq   = (const float*)d_in[3];
  const float* wk   = (const float*)d_in[4];
  const float* bk   = (const float*)d_in[5];
  const float* wv   = (const float*)d_in[6];
  const float* bv   = (const float*)d_in[7];
  const float* wo   = (const float*)d_in[8];
  const float* bo   = (const float*)d_in[9];

  char* ws = (char*)d_ws;
  size_t off = 0;
  auto take = [&](size_t bytes) -> void* {
    void* p = ws + off;
    off += (bytes + 255) & ~(size_t)255;
    return p;
  };
  __bf16* xb   = (__bf16*)take((size_t)kBS * kE * 2);          // 8 MiB
  __bf16* wqT  = (__bf16*)take((size_t)kH * kDH * kE * 2);     // 2 MiB
  __bf16* wkT  = (__bf16*)take((size_t)kH * kDH * kE * 2);
  __bf16* wvT  = (__bf16*)take((size_t)kH * kDH * kE * 2);
  __bf16* woT  = (__bf16*)take((size_t)kE * kE * 2);           // 2 MiB
  __bf16* qg   = (__bf16*)take((size_t)kB * kH * kS * kDH * 2);// 8 MiB
  __bf16* kgp  = (__bf16*)take((size_t)kB * kH * kS * kDH * 2);
  __bf16* vtg  = (__bf16*)take((size_t)kB * kH * kDH * kS * 2);
  __bf16* obuf = (__bf16*)take((size_t)kBS * kE * 2);          // 8 MiB

  const int nX = kBS * kE;
  cvt_f32_bf16<<<(nX + 255) / 256, 256, 0, stream>>>(x, xb, nX);
  const int nW = kH * kE * kDH;
  cvt_transpose_head_w<<<(nW + 255) / 256, 256, 0, stream>>>(wq, wqT);
  cvt_transpose_head_w<<<(nW + 255) / 256, 256, 0, stream>>>(wk, wkT);
  cvt_transpose_head_w<<<(nW + 255) / 256, 256, 0, stream>>>(wv, wvT);
  const int nWo = kE * kE;
  cvt_transpose_wo<<<(nWo + 255) / 256, 256, 0, stream>>>(wo, woT);

  dim3 gq(kBS / 32, kH);
  qkv_proj<<<gq, 32, 0, stream>>>(xb, wqT, bq, qg, 0, 0.125f);  // q pre-scaled
  qkv_proj<<<gq, 32, 0, stream>>>(xb, wkT, bk, kgp, 0, 1.0f);
  qkv_proj<<<gq, 32, 0, stream>>>(xb, wvT, bv, vtg, 1, 1.0f);   // v transposed

  attn_flash<<<dim3(kS / 64, kH, kB), 128, 0, stream>>>(qg, kgp, vtg, mask,
                                                        obuf);

  out_proj<<<dim3(kBS / 32, kE / 64), 32, 0, stream>>>(obuf, woT, bo,
                                                       (float*)d_out);
}
